// BaseRNN_88029649699270
// MI455X (gfx1250) — matrix operand order repaired and checked by reading, not compiled, not run on hardware
//
#include <hip/hip_runtime.h>
#include <hip/hip_bf16.h>

typedef __attribute__((ext_vector_type(16))) __bf16 v16bf;
typedef __attribute__((ext_vector_type(8)))  __bf16 v8bf;
typedef __attribute__((ext_vector_type(8)))  float  v8f;

#define NBLK 64
#define NTHR 256

constexpr int Bb  = 256;     // batch
constexpr int Tt  = 1024;    // timesteps
constexpr int Dd  = 64;      // input dim
constexpr int Hh  = 512;     // hidden
constexpr int Oo  = 128;     // decoder out (2*D)
constexpr int KX  = 96;      // padded [t | x] columns (1+64 -> 96, K%32==0)
constexpr int KC  = KX + Hh; // 608 combined K for r/z GEMM
constexpr int NRZ = 1024;    // r,z output columns

// ---------------- WMMA helpers (layouts per CDNA5 ISA 7.12.2) ----------------

__device__ __forceinline__ v8f wmma_bf16(v16bf a, v16bf b, v8f c) {
  return __builtin_amdgcn_wmma_f32_16x16x32_bf16(false, a, false, b, (short)0, c,
                                                 false, false);
}

// A fragment from a per-lane base pointer (already includes row & half-wave
// swizzle): elems 0..7 at p[0..7], elems 8..15 at p[16..23].
__device__ __forceinline__ v16bf ld_a(const __hip_bfloat16* p) {
  v8bf lo = *reinterpret_cast<const v8bf*>(p);
  v8bf hi = *reinterpret_cast<const v8bf*>(p + 16);
  v16bf a;
#pragma unroll
  for (int i = 0; i < 8; ++i) { a[i] = lo[i]; a[i + 8] = hi[i]; }
  return a;
}

// B fragment: elems 0..7 at p[0..7], elems 8..15 at p[16..23] (same as A).
__device__ __forceinline__ v16bf ld_b(const __hip_bfloat16* p) {
  return ld_a(p);
}

// 32x32 output tile: all pointers pre-swizzled per lane, K/LDG compile-time.
template <int K, int LDG>
__device__ __forceinline__ void tile32(const __hip_bfloat16* pa0,
                                       const __hip_bfloat16* pa1,
                                       const __hip_bfloat16* pb0,
                                       const __hip_bfloat16* pb1,
                                       float* g0, float* g1) {
  v8f a00 = {}, a01 = {}, a10 = {}, a11 = {};
#pragma unroll 4
  for (int k0 = 0; k0 < K; k0 += 32) {
    v16bf A0 = ld_a(pa0 + k0);
    v16bf A1 = ld_a(pa1 + k0);
    v16bf B0 = ld_b(pb0 + k0);
    v16bf B1 = ld_b(pb1 + k0);
    a00 = wmma_bf16(A0, B0, a00);
    a01 = wmma_bf16(A0, B1, a01);
    a10 = wmma_bf16(A1, B0, a10);
    a11 = wmma_bf16(A1, B1, a11);
  }
#pragma unroll
  for (int i = 0; i < 8; ++i) {
    g0[0] = a00[i]; g0[16] = a01[i];
    g1[0] = a10[i]; g1[16] = a11[i];
    g0 += LDG; g1 += LDG;
  }
}

// Grid-wide barrier: monotonic counter (zeroed by prep kernel each call).
__device__ __forceinline__ void gsync(unsigned* cnt, unsigned& bars) {
  __syncthreads();
  if (threadIdx.x == 0) {
    ++bars;
    __threadfence();
    atomicAdd(cnt, 1u);
    const unsigned tgt = bars * NBLK;
    while (__hip_atomic_load(cnt, __ATOMIC_ACQUIRE, __HIP_MEMORY_SCOPE_AGENT) < tgt)
      __builtin_amdgcn_s_sleep(1);
  }
  __syncthreads();
}

// ---------------- prep: bf16 weight packing, state init, sync reset ----------

__global__ void gru_prep(unsigned* sync, __hip_bfloat16* A_cat,
                         __hip_bfloat16* W_rz, __hip_bfloat16* W_inn,
                         __hip_bfloat16* W_hhn, __hip_bfloat16* W_decb,
                         float* hbuf,
                         const float* xs, const float* ts,
                         const float* W_ih, const float* W_hh,
                         const float* W_dec) {
  const long nWrz  = (long)NRZ * KC;
  const long nWin  = (long)Hh * KX;
  const long nWhh  = (long)Hh * Hh;
  const long nWdec = (long)Oo * Hh;
  const long nA    = (long)Bb * KC;
  const long nH    = (long)Bb * Hh;
  const long nS    = 64;
  const long total = nWrz + nWin + nWhh + nWdec + nA + nH + nS;
  const long stride = (long)gridDim.x * blockDim.x;
  for (long i = (long)blockIdx.x * blockDim.x + threadIdx.x; i < total; i += stride) {
    long idx = i;
    if (idx < nWrz) {
      int n = (int)(idx / KC), k = (int)(idx % KC);
      float v = (k < 65) ? W_ih[(long)n * 65 + k]
              : (k < KX) ? 0.f
                         : W_hh[(long)n * Hh + (k - KX)];
      W_rz[idx] = __float2bfloat16(v);
      continue;
    }
    idx -= nWrz;
    if (idx < nWin) {
      int n = (int)(idx / KX), k = (int)(idx % KX);
      float v = (k < 65) ? W_ih[(long)(1024 + n) * 65 + k] : 0.f;
      W_inn[idx] = __float2bfloat16(v);
      continue;
    }
    idx -= nWin;
    if (idx < nWhh) { W_hhn[idx] = __float2bfloat16(W_hh[(long)1024 * Hh + idx]); continue; }
    idx -= nWhh;
    if (idx < nWdec) { W_decb[idx] = __float2bfloat16(W_dec[idx]); continue; }
    idx -= nWdec;
    if (idx < nA) {
      int b = (int)(idx / KC), k = (int)(idx % KC);
      float v;
      if (k == 0)      v = ts[(long)b * Tt];
      else if (k < 65) v = xs[(long)b * Tt * Dd + (k - 1)];
      else             v = 0.f;
      A_cat[idx] = __float2bfloat16(v);
      continue;
    }
    idx -= nA;
    if (idx < nH) { hbuf[idx] = 0.f; continue; }
    idx -= nH;
    sync[idx] = 0u;
  }
}

// ---------------- persistent GRU scan --------------------------------------

__global__ __launch_bounds__(NTHR, 1)
void gru_persistent(__hip_bfloat16* A_cat,
                    const __hip_bfloat16* W_rz, const __hip_bfloat16* W_inn,
                    const __hip_bfloat16* W_hhn, const __hip_bfloat16* W_decb,
                    float* g_rz, float* g_in, float* g_hn, float* hbuf,
                    unsigned* sync,
                    const float* xs, const float* ts,
                    const float* b_ih, const float* b_hh, const float* b_dec,
                    float* out, const int* cutoff_ptr) {
  const int lane = threadIdx.x & 31;
  const int wv   = threadIdx.x >> 5;
  const int bid  = blockIdx.x;
  const int w    = bid * 8 + wv;
  const int r    = lane & 15;       // row / N-col within fragment
  const int hf   = lane >> 4;       // half-wave select
  const int cutoff = *cutoff_ptr;
  unsigned bars = 0;

  // -------- Phase-1 tile pointers: invariant across t, hoisted out. --------
  int role, m0, n0;
  const __hip_bfloat16 *pa0, *pa1, *pb0, *pb1;
  float *pg0, *pg1;
  if (w < 256) {                 // r,z combined: (256x608)x(608x1024)
    role = 0; m0 = (w >> 5) * 32; n0 = (w & 31) * 32;
    pa0 = A_cat + (m0 + r) * KC + hf * 8;
    pb0 = W_rz + (n0 + r) * KC + hf * 8;
    pg0 = g_rz + (m0 + hf * 8) * NRZ + n0 + r;
    pa1 = pa0 + 16 * KC; pb1 = pb0 + 16 * KC; pg1 = pg0 + 16 * NRZ;
  } else if (w < 384) {          // h_n: (256x512)x(512x512)
    int v = w - 256;
    role = 1; m0 = (v >> 4) * 32; n0 = (v & 15) * 32;
    pa0 = A_cat + KX + (m0 + r) * KC + hf * 8;
    pb0 = W_hhn + (n0 + r) * Hh + hf * 8;
    pg0 = g_hn + (m0 + hf * 8) * Hh + n0 + r;
    pa1 = pa0 + 16 * KC; pb1 = pb0 + 16 * Hh; pg1 = pg0 + 16 * Hh;
  } else {                       // i_n: (256x96)x(96x512)
    int v = w - 384;
    role = 2; m0 = (v >> 4) * 32; n0 = (v & 15) * 32;
    pa0 = A_cat + (m0 + r) * KC + hf * 8;
    pb0 = W_inn + (n0 + r) * KX + hf * 8;
    pg0 = g_in + (m0 + hf * 8) * Hh + n0 + r;
    pa1 = pa0 + 16 * KC; pb1 = pb0 + 16 * KX; pg1 = pg0 + 16 * Hh;
  }

  // -------- Phase-3 decoder pointers (blocks 0..7), also t-invariant. ------
  const int r0  = bid * 32;       // batch rows owned by this block
  const int nd  = wv * 16 + r;    // decoder output column for this lane
  const __hip_bfloat16* dpa0 = A_cat + KX + (r0 + r) * KC + hf * 8;
  const __hip_bfloat16* dpa1 = dpa0 + 16 * KC;
  const __hip_bfloat16* dpb  = W_decb + nd * Hh + hf * 8;

  for (int t = 0; t < Tt; ++t) {
    // ---- Phase 1: one 32x32 gate-GEMM tile per wave (512 tiles). ----
    if (role == 0)      tile32<KC, NRZ>(pa0, pa1, pb0, pb1, pg0, pg1);
    else if (role == 1) tile32<Hh, Hh>(pa0, pa1, pb0, pb1, pg0, pg1);
    else                tile32<KX, Hh>(pa0, pa1, pb0, pb1, pg0, pg1);
    gsync(sync, bars);

    // ---- Phases 2+3: blocks 0..7 each own 32 batch rows end-to-end. ----
    if (bid < 8) {
      // Phase 2: gate fusion -> h_new (f32 state + bf16 copy into A_cat)
      for (int idx = threadIdx.x; idx < 32 * Hh; idx += NTHR) {
        const int br = r0 + (idx >> 9);
        const int j  = idx & 511;
        float gr = g_rz[br * NRZ + j]       + b_ih[j]        + b_hh[j];
        float gz = g_rz[br * NRZ + 512 + j] + b_ih[512 + j]  + b_hh[512 + j];
        float gi = g_in[br * Hh + j]        + b_ih[1024 + j];
        float gh = g_hn[br * Hh + j]        + b_hh[1024 + j];
        float rg = 1.f / (1.f + expf(-gr));
        float zg = 1.f / (1.f + expf(-gz));
        float ng = tanhf(gi + rg * gh);
        float hold = hbuf[br * Hh + j];
        float hnew = (1.f - zg) * ng + zg * hold;
        hbuf[br * Hh + j] = hnew;
        A_cat[br * KC + KX + j] = __float2bfloat16(hnew);
      }
      __threadfence();
      __syncthreads();

      // Phase 3: decoder GEMM (32 rows x 128 cols), wave wv owns 16 cols.
      v8f c0 = {}, c1 = {};
#pragma unroll 4
      for (int k0 = 0; k0 < Hh; k0 += 32) {
        v16bf A0 = ld_a(dpa0 + k0);
        v16bf A1 = ld_a(dpa1 + k0);
        v16bf Bf = ld_b(dpb + k0);
        c0 = wmma_bf16(A0, Bf, c0);
        c1 = wmma_bf16(A1, Bf, c1);
      }
      const float bd = b_dec[nd];
      const bool has_next   = (t + 1) < Tt;
      const bool force_next = (t + 1) < cutoff;
#pragma unroll
      for (int mi = 0; mi < 2; ++mi) {
        const v8f& c = mi ? c1 : c0;
#pragma unroll
        for (int i = 0; i < 8; ++i) {
          const int m = r0 + mi * 16 + hf * 8 + i;     // batch row
          float val = c[i] + bd;
          // 32-bit index: ((m*Tt + t)*Oo + nd) <= 33.5M, mults are shifts
          const int oidx = ((m << 10) + t) * Oo + nd;
          if (nd < Dd) {
            out[oidx] = val;                            // mean
            if (has_next) {                             // next step's x input
              float nx = force_next ? xs[((m << 10) + (t + 1)) * Dd + nd] : val;
              A_cat[m * KC + 1 + nd] = __float2bfloat16(nx);
              if (nd == 0)
                A_cat[m * KC] = __float2bfloat16(ts[(m << 10) + (t + 1)]);
            }
          } else {
            out[oidx] = fmaxf(val, 1e-3f);              // std clamp
          }
        }
      }
    }
    gsync(sync, bars);
  }
}

// ---------------- host launch ----------------------------------------------

extern "C" void kernel_launch(void* const* d_in, const int* in_sizes, int n_in,
                              void* d_out, int out_size, void* d_ws, size_t ws_size,
                              hipStream_t stream) {
  const float* xs    = (const float*)d_in[0];
  const float* ts    = (const float*)d_in[1];
  const float* W_ih  = (const float*)d_in[2];
  const float* W_hh  = (const float*)d_in[3];
  const float* b_ih  = (const float*)d_in[4];
  const float* b_hh  = (const float*)d_in[5];
  const float* W_dec = (const float*)d_in[6];
  const float* b_dec = (const float*)d_in[7];
  const int*   cut   = (const int*)d_in[8];
  float* out = (float*)d_out;

  char* p = (char*)d_ws;
  unsigned*        sync   = (unsigned*)p;        p += 256;
  __hip_bfloat16*  A_cat  = (__hip_bfloat16*)p;  p += (size_t)Bb * KC * 2;
  __hip_bfloat16*  W_rz   = (__hip_bfloat16*)p;  p += (size_t)NRZ * KC * 2;
  __hip_bfloat16*  W_inn  = (__hip_bfloat16*)p;  p += (size_t)Hh * KX * 2;
  __hip_bfloat16*  W_hhn  = (__hip_bfloat16*)p;  p += (size_t)Hh * Hh * 2;
  __hip_bfloat16*  W_decb = (__hip_bfloat16*)p;  p += (size_t)Oo * Hh * 2;
  float*           g_rz   = (float*)p;           p += (size_t)Bb * NRZ * 4;
  float*           g_in   = (float*)p;           p += (size_t)Bb * Hh * 4;
  float*           g_hn   = (float*)p;           p += (size_t)Bb * Hh * 4;
  float*           hbuf   = (float*)p;           p += (size_t)Bb * Hh * 4;

  gru_prep<<<512, NTHR, 0, stream>>>(sync, A_cat, W_rz, W_inn, W_hhn, W_decb,
                                     hbuf, xs, ts, W_ih, W_hh, W_dec);
  gru_persistent<<<NBLK, NTHR, 0, stream>>>(A_cat, W_rz, W_inn, W_hhn, W_decb,
                                            g_rz, g_in, g_hn, hbuf, sync,
                                            xs, ts, b_ih, b_hh, b_dec, out, cut);
}
